// CausalSelfAttention_28613072126594
// MI455X (gfx1250) — matrix operand order repaired and checked by reading, not compiled, not run on hardware
//
#include <hip/hip_runtime.h>
#include <stdint.h>

// CDNA5 / gfx1250 wave32 WMMA implementation of causal self-attention.
// Precision: bf16 inputs to v_wmma_f32_16x16x32_bf16, fp32 accumulation.
// Attention kernel stages V tiles through LDS with GLOBAL_LOAD_ASYNC_TO_LDS
// (per-wave ASYNCcnt double-duty as a prefetch: issued at top of each
// kk-step, consumed after the score WMMAs + softmax).
// Workspace layout (bytes):
//   [0,16M)        x in bf16            (8192x1024)
//   [16M,24M)      Wq,Wk,Wv,Wp in bf16  (4 x 1024x1024)
//   [24M,40M)      Q bf16 [B,H,S,64]    (scaled by 1/8)
//   [40M,56M)      K bf16 [B,H,S,64]
//   [56M,72M)      V^T bf16 [B,H,64,S]
//   [72M,88M)      attn-out bf16 [B*S, 1024]

typedef __attribute__((ext_vector_type(16))) __bf16 v16bf;
typedef __attribute__((ext_vector_type(8)))  float  v8f;

#define BSZ 4
#define SSZ 2048
#define DSZ 1024
#define HN  16
#define HDM 64
#define MSZ (BSZ*SSZ) /* 8192 */

union FragB16 { v16bf v; uint4 q[2]; };

__device__ __forceinline__ __bf16 f2bf(float f) {
  unsigned u = __float_as_uint(f);
  u += 0x7fffu + ((u >> 16) & 1u);          // round-to-nearest-even
  unsigned short h = (unsigned short)(u >> 16);
  __bf16 r; __builtin_memcpy(&r, &h, 2);
  return r;
}

__global__ void cast_f32_to_bf16(const float* __restrict__ in,
                                 __bf16* __restrict__ out, int n) {
  int i = blockIdx.x * blockDim.x + threadIdx.x;
  if (i < n) out[i] = f2bf(in[i]);
}

// ---------------------------------------------------------------------------
// GEMM: out = A[M,K] * W[N,K]^T, M=8192, N=K=1024.
// Block = 8 waves (4 in M x 2 in N); wave computes 32x64 as 2x4 WMMA tiles.
// mode 0: Q store [B,H,S,64] scaled by 0.125; mode 1: K store [B,H,S,64];
// mode 2: V store transposed [B,H,64,S].
// ---------------------------------------------------------------------------
__global__ __launch_bounds__(256)
void gemm_qkv(const __bf16* __restrict__ A, const __bf16* __restrict__ W,
              __bf16* __restrict__ out, int mode) {
  const int lane = threadIdx.x & 31;
  const int wave = threadIdx.x >> 5;
  const int sub  = lane >> 4;      // 0/1: which half of the wave
  const int r    = lane & 15;
  const int m0 = blockIdx.x * 128 + (wave >> 1) * 32;
  const int n0 = blockIdx.y * 128 + (wave & 1) * 64;

  v8f acc[2][4];
  for (int i = 0; i < 2; ++i)
    for (int j = 0; j < 4; ++j)
      for (int e = 0; e < 8; ++e) acc[i][j][e] = 0.f;

  for (int k0 = 0; k0 < DSZ; k0 += 32) {
    FragB16 a[2];
    for (int i = 0; i < 2; ++i) {
      const __bf16* p = A + (size_t)(m0 + i * 16 + r) * DSZ + k0 + sub * 8;
      a[i].q[0] = *(const uint4*)p;
      a[i].q[1] = *(const uint4*)(p + 16);
    }
    for (int j = 0; j < 4; ++j) {
      FragB16 b;
      const __bf16* p = W + (size_t)(n0 + j * 16 + r) * DSZ + k0 + sub * 8;
      b.q[0] = *(const uint4*)p;
      b.q[1] = *(const uint4*)(p + 16);
      for (int i = 0; i < 2; ++i)
        acc[i][j] = __builtin_amdgcn_wmma_f32_16x16x32_bf16(
            false, a[i].v, false, b.v, (short)0, acc[i][j], false, false);
    }
  }

  const float qscale = (mode == 0) ? 0.125f : 1.0f;  // fold HD^-0.5 into Q
  for (int i = 0; i < 2; ++i)
    for (int j = 0; j < 4; ++j)
      for (int v = 0; v < 8; ++v) {
        int gm = m0 + i * 16 + sub * 8 + v;
        int gn = n0 + j * 16 + r;
        int b = gm >> 11, s = gm & (SSZ - 1);
        int h = gn >> 6,  d = gn & (HDM - 1);
        float val = acc[i][j][v] * qscale;
        if (mode == 2)
          out[((size_t)((b * HN + h) * HDM + d)) * SSZ + s] = f2bf(val);
        else
          out[((size_t)((b * HN + h) * SSZ + s)) * HDM + d] = f2bf(val);
      }
}

// ---------------------------------------------------------------------------
// Flash attention: one wave owns a 16-row Q tile; streams K/V in 32-col steps
// with online softmax. Causal mask is a VALU select (EXEC stays all-ones).
// V fragments are prefetched with GLOBAL_LOAD_ASYNC_TO_LDS_B128 (ASYNCcnt)
// at the top of each step and consumed after scores+softmax.
// P (C-layout) -> A-layout transpose goes through per-wave LDS.
// ---------------------------------------------------------------------------
__global__ __launch_bounds__(256)
void attn_fwd(const __bf16* __restrict__ Q, const __bf16* __restrict__ K,
              const __bf16* __restrict__ Vt, __bf16* __restrict__ AO) {
  __shared__ __align__(16) __bf16 plds[8][16][40];     // per-wave 16x32 (+pad)
  __shared__ __align__(16) char   vstage[8][4096];     // per-wave V staging
  const int lane = threadIdx.x & 31;
  const int wave = threadIdx.x >> 5;
  const int sub  = lane >> 4;
  const int r    = lane & 15;
  const int w  = blockIdx.x * 8 + wave;          // 8192 wave tasks
  const int qt = w & (SSZ / 16 - 1);             // 128 q-tiles per head
  const int bh = w >> 7;                         // 64 (b,h) pairs
  const int q0 = qt * 16;
  const __bf16* Qp = Q  + (size_t)bh * SSZ * HDM;
  const __bf16* Kp = K  + (size_t)bh * SSZ * HDM;
  const __bf16* Vp = Vt + (size_t)bh * HDM * SSZ;

  // wave-relative LDS byte offsets of this lane's 8 V staging slots
  unsigned vslot[8];
  for (int j = 0; j < 4; ++j)
    for (int c = 0; c < 2; ++c)
      vslot[j * 2 + c] =
          (unsigned)(uintptr_t)&vstage[wave][(j * 2 + c) * 512 + lane * 16];

  FragB16 qa[2];                                 // Q A-frags, d 0..31 / 32..63
  for (int hs = 0; hs < 2; ++hs) {
    const __bf16* p = Qp + (size_t)(q0 + r) * HDM + hs * 32 + sub * 8;
    qa[hs].q[0] = *(const uint4*)p;
    qa[hs].q[1] = *(const uint4*)(p + 16);
  }

  v8f o[4];
  for (int j = 0; j < 4; ++j)
    for (int e = 0; e < 8; ++e) o[j][e] = 0.f;
  float mrow[8], lrow[8];
  for (int v = 0; v < 8; ++v) { mrow[v] = -1e30f; lrow[v] = 0.f; }

  const int nsteps = (q0 + 47) >> 5;             // cover kk <= q0+15
  for (int st = 0; st < nsteps; ++st) {
    const int kk0 = st * 32;

    // --- async prefetch of this step's V fragments into LDS (ASYNCcnt) ---
    // make sure last step's ds reads of these slots are done before rewrite
    asm volatile("s_wait_dscnt 0" ::: "memory");
    for (int j = 0; j < 4; ++j)
      for (int c = 0; c < 2; ++c) {
        const __bf16* g = Vp + (size_t)(j * 16 + r) * SSZ + kk0 + sub * 8 + c * 16;
        asm volatile("global_load_async_to_lds_b128 %0, %1, off"
                     :: "v"(vslot[j * 2 + c]), "v"(g) : "memory");
      }

    // --- scores: S = Q K^T (head dim reduced as 2x32) ---
    v8f s0, s1;
    for (int e = 0; e < 8; ++e) { s0[e] = 0.f; s1[e] = 0.f; }
    for (int hs = 0; hs < 2; ++hs) {
      FragB16 kb0, kb1;
      const __bf16* p0 = Kp + (size_t)(kk0 + r) * HDM + hs * 32 + sub * 8;
      kb0.q[0] = *(const uint4*)p0;  kb0.q[1] = *(const uint4*)(p0 + 16);
      const __bf16* p1 = Kp + (size_t)(kk0 + 16 + r) * HDM + hs * 32 + sub * 8;
      kb1.q[0] = *(const uint4*)p1;  kb1.q[1] = *(const uint4*)(p1 + 16);
      s0 = __builtin_amdgcn_wmma_f32_16x16x32_bf16(false, qa[hs].v, false, kb0.v,
                                                   (short)0, s0, false, false);
      s1 = __builtin_amdgcn_wmma_f32_16x16x32_bf16(false, qa[hs].v, false, kb1.v,
                                                   (short)0, s1, false, false);
    }

    // --- causal mask + online softmax (row = 16 lanes of one half-wave) ---
    float p0v[8], p1v[8], alpha[8];
    for (int v = 0; v < 8; ++v) {
      const int qrow = q0 + sub * 8 + v;
      float x0 = ((kk0 + r)      <= qrow) ? s0[v] : -1e30f;
      float x1 = ((kk0 + 16 + r) <= qrow) ? s1[v] : -1e30f;
      float rm = fmaxf(x0, x1);
      for (int off = 1; off < 16; off <<= 1)
        rm = fmaxf(rm, __shfl_xor(rm, off, 32));
      float mnew = fmaxf(mrow[v], rm);
      float al = __expf(mrow[v] - mnew);
      float e0 = __expf(x0 - mnew);
      float e1 = __expf(x1 - mnew);
      float rs = e0 + e1;
      for (int off = 1; off < 16; off <<= 1)
        rs += __shfl_xor(rs, off, 32);
      lrow[v] = lrow[v] * al + rs;
      mrow[v] = mnew;
      alpha[v] = al;
      p0v[v] = e0;  p1v[v] = e1;
    }
    for (int j = 0; j < 4; ++j)
      for (int v = 0; v < 8; ++v) o[j][v] *= alpha[v];

    // --- C-layout -> A-layout transpose of P through this wave's LDS ---
    for (int v = 0; v < 8; ++v) {
      plds[wave][sub * 8 + v][r]      = f2bf(p0v[v]);
      plds[wave][sub * 8 + v][16 + r] = f2bf(p1v[v]);
    }
    asm volatile("s_wait_dscnt 0" ::: "memory");   // same-wave DS RAW
    FragB16 pa;
    pa.q[0] = *(const uint4*)&plds[wave][r][sub * 8];
    pa.q[1] = *(const uint4*)&plds[wave][r][16 + sub * 8];

    // --- consume prefetched V: O += P @ V (4 d-tiles) ---
    asm volatile("s_wait_asynccnt 0" ::: "memory");
    for (int j = 0; j < 4; ++j) {
      FragB16 vb;
      vb.q[0] = *(const uint4*)&vstage[wave][(j * 2 + 0) * 512 + lane * 16];
      vb.q[1] = *(const uint4*)&vstage[wave][(j * 2 + 1) * 512 + lane * 16];
      o[j] = __builtin_amdgcn_wmma_f32_16x16x32_bf16(false, pa.v, false, vb.v,
                                                     (short)0, o[j], false, false);
    }
  }

  const int b = bh >> 4, h = bh & 15;
  for (int j = 0; j < 4; ++j)
    for (int v = 0; v < 8; ++v) {
      int row = b * SSZ + q0 + sub * 8 + v;
      int col = h * HDM + j * 16 + r;
      AO[(size_t)row * DSZ + col] = f2bf(o[j][v] / lrow[v]);
    }
}

// ---------------------------------------------------------------------------
// Output projection: y = AO * Wp^T + bp, fp32 output.
// ---------------------------------------------------------------------------
__global__ __launch_bounds__(256)
void gemm_out(const __bf16* __restrict__ A, const __bf16* __restrict__ W,
              const float* __restrict__ bias, float* __restrict__ out) {
  const int lane = threadIdx.x & 31;
  const int wave = threadIdx.x >> 5;
  const int sub  = lane >> 4;
  const int r    = lane & 15;
  const int m0 = blockIdx.x * 128 + (wave >> 1) * 32;
  const int n0 = blockIdx.y * 128 + (wave & 1) * 64;

  v8f acc[2][4];
  for (int i = 0; i < 2; ++i)
    for (int j = 0; j < 4; ++j)
      for (int e = 0; e < 8; ++e) acc[i][j][e] = 0.f;

  for (int k0 = 0; k0 < DSZ; k0 += 32) {
    FragB16 a[2];
    for (int i = 0; i < 2; ++i) {
      const __bf16* p = A + (size_t)(m0 + i * 16 + r) * DSZ + k0 + sub * 8;
      a[i].q[0] = *(const uint4*)p;
      a[i].q[1] = *(const uint4*)(p + 16);
    }
    for (int j = 0; j < 4; ++j) {
      FragB16 b;
      const __bf16* p = W + (size_t)(n0 + j * 16 + r) * DSZ + k0 + sub * 8;
      b.q[0] = *(const uint4*)p;
      b.q[1] = *(const uint4*)(p + 16);
      for (int i = 0; i < 2; ++i)
        acc[i][j] = __builtin_amdgcn_wmma_f32_16x16x32_bf16(
            false, a[i].v, false, b.v, (short)0, acc[i][j], false, false);
    }
  }

  for (int i = 0; i < 2; ++i)
    for (int j = 0; j < 4; ++j)
      for (int v = 0; v < 8; ++v) {
        int gm = m0 + i * 16 + sub * 8 + v;
        int gn = n0 + j * 16 + r;
        out[(size_t)gm * DSZ + gn] = acc[i][j][v] + bias[gn];
      }
}

extern "C" void kernel_launch(void* const* d_in, const int* in_sizes, int n_in,
                              void* d_out, int out_size, void* d_ws, size_t ws_size,
                              hipStream_t stream) {
  const float* x  = (const float*)d_in[0];
  const float* Wq = (const float*)d_in[1];
  const float* Wk = (const float*)d_in[2];
  const float* Wv = (const float*)d_in[3];
  const float* Wp = (const float*)d_in[4];
  const float* bp = (const float*)d_in[5];
  float* out = (float*)d_out;
  char* ws = (char*)d_ws;

  const size_t NX = (size_t)MSZ * DSZ;   // 8388608
  const size_t NW = (size_t)DSZ * DSZ;   // 1048576

  __bf16* xb  = (__bf16*)(ws);
  __bf16* wqb = (__bf16*)(ws + 16777216);
  __bf16* wkb = (__bf16*)(ws + 16777216 + 2097152);
  __bf16* wvb = (__bf16*)(ws + 16777216 + 2 * 2097152);
  __bf16* wpb = (__bf16*)(ws + 16777216 + 3 * 2097152);
  __bf16* Qb  = (__bf16*)(ws + 25165824);
  __bf16* Kb  = (__bf16*)(ws + 25165824 + 16777216);
  __bf16* Vtb = (__bf16*)(ws + 25165824 + 2 * 16777216);
  __bf16* AOb = (__bf16*)(ws + 25165824 + 3 * 16777216);

  cast_f32_to_bf16<<<(int)(NX / 256), 256, 0, stream>>>(x,  xb,  (int)NX);
  cast_f32_to_bf16<<<(int)(NW / 256), 256, 0, stream>>>(Wq, wqb, (int)NW);
  cast_f32_to_bf16<<<(int)(NW / 256), 256, 0, stream>>>(Wk, wkb, (int)NW);
  cast_f32_to_bf16<<<(int)(NW / 256), 256, 0, stream>>>(Wv, wvb, (int)NW);
  cast_f32_to_bf16<<<(int)(NW / 256), 256, 0, stream>>>(Wp, wpb, (int)NW);

  dim3 gg(MSZ / 128, DSZ / 128);  // 64 x 8
  gemm_qkv<<<gg, 256, 0, stream>>>(xb, wqb, Qb, 0);
  gemm_qkv<<<gg, 256, 0, stream>>>(xb, wkb, Kb, 1);
  gemm_qkv<<<gg, 256, 0, stream>>>(xb, wvb, Vtb, 2);

  attn_fwd<<<(BSZ * HN * (SSZ / 16)) / 8, 256, 0, stream>>>(Qb, Kb, Vtb, AOb);

  gemm_out<<<gg, 256, 0, stream>>>(AOb, wpb, bp, out);
}
